// MyModel_73237782331570
// MI455X (gfx1250) — compile-verified, matrix operand order checked by reading, not run on hardware
//
#include <hip/hip_runtime.h>
#include <hip/hip_bf16.h>
#include <float.h>

// ---------------------------------------------------------------------------
// HierBiLSTM doc classifier for MI455X (gfx1250, wave32, WMMA).
// bf16 A/B matrices, f32 accumulation (v_wmma_f32_16x16x32_bf16).
// Word recurrence: persistent-RNN, Whh B-fragments register-resident
// (20 waves/block, <=2 N-tiles per wave, dual accumulators sharing one
// A-fragment -> live set ~104 VGPRs, no spills), scalar tile guards.
// ---------------------------------------------------------------------------

typedef __bf16 bf16_t;
typedef __attribute__((ext_vector_type(16))) __bf16 v16bf;
typedef __attribute__((ext_vector_type(8)))  float  v8f;

#define S_DOC  1024
#define W_SENT 128
#define D_EMB  300
#define H_HID  150
#define NG     600      // 4*H gates
#define NT     38       // N tiles of 16 covering 608
#define NPAD   608
#define KC_IN  10       // k-chunks of 32 covering K=300 (pad 320)
#define KC_H   5        // k-chunks of 32 covering K=150 (pad 160)
#define C_LBL  7
#define WL_WAVES   20
#define WL_THREADS (WL_WAVES * 32)   // 640: <=2 N-tiles per wave

// 16-bit A/B fragment K mapping (per lane, element j of 16):
//   lanes 0-15: koff=0, lanes 16-31: koff=8
//   v=j>>1, p=j&1 : k = v<4 ? koff+2v+p : 16+koff+2(v-4)+p
__device__ __forceinline__ int frag_k(int j, int koff) {
    int v = j >> 1, p = j & 1;
    return (v < 4) ? (koff + v * 2 + p) : (16 + koff + (v - 4) * 2 + p);
}

__device__ __forceinline__ float sigf(float x) {
    return 1.0f / (1.0f + __expf(-x));
}

// ---------------------------------------------------------------------------
// Pack a [NG x Kv] f32 weight into WMMA-B-fragment order:
// dst[nt][kc][lane][16] bf16; element (k,n), n = nt*16 + (lane&15),
// k = kc*32 + frag_k(j, (lane>>4)*8).
// ---------------------------------------------------------------------------
__global__ void pack_b_kernel(const float* __restrict__ Wt, int Kv, int kchunks,
                              bf16_t* __restrict__ dst) {
    const int nt = blockIdx.x, kc = blockIdx.y, lane = threadIdx.x;
    const int n = nt * 16 + (lane & 15);
    const int koff = (lane >> 4) * 8;
    const size_t base = ((size_t)(nt * kchunks + kc)) * 512 + (size_t)lane * 16;
#pragma unroll
    for (int j = 0; j < 16; ++j) {
        int k = kc * 32 + frag_k(j, koff);
        float v = (n < NG && k < Kv) ? Wt[(size_t)n * Kv + k] : 0.0f;
        dst[base + j] = (bf16_t)v;
    }
}

// ---------------------------------------------------------------------------
// Input-projection GEMM: Out[dir][row][n] = A[row,:300] @ W.T + (bih+bhh).
// Each wave holds its 16x320 A tile in registers and sweeps all 38 N-tiles;
// B fragments are double-buffered so v_wmma overlaps the next global load.
// TRANS=true scatters word-level output time-major: row s*128+t -> [t][s].
// grid=(Mtiles/8, 1, 2 dirs), block=256 (8 waves).
// ---------------------------------------------------------------------------
template <bool OUT_BF16, bool TRANS>
__global__ void inproj_gemm_kernel(const float* __restrict__ A, long Mtiles, long Mrows,
                                   const bf16_t* __restrict__ packedB,   // [2][NT][KC_IN][512]
                                   const float* __restrict__ bih0, const float* __restrict__ bhh0,
                                   const float* __restrict__ bih1, const float* __restrict__ bhh1,
                                   void* __restrict__ outp) {
    const int lane = threadIdx.x & 31;
    const int wid  = __builtin_amdgcn_readfirstlane(threadIdx.x >> 5);
    const int dir  = blockIdx.z;
    const long mt  = (long)blockIdx.x * 8 + wid;
    if (mt >= Mtiles) return;

    const int m    = lane & 15;
    const int half = lane >> 4;
    const int koff = half * 8;
    const float* bih = dir ? bih1 : bih0;
    const float* bhh = dir ? bhh1 : bhh0;
    const long row0 = mt * 16;

    // Gather the 16x320 A tile as bf16 fragments (K >= 300 zero padded).
    v16bf afr[KC_IN];
#pragma unroll
    for (int kc = 0; kc < KC_IN; ++kc) {
        if (kc + 1 < KC_IN)
            __builtin_prefetch(&A[(row0 + m) * (long)D_EMB + (kc + 1) * 32], 0, 3);
#pragma unroll
        for (int j = 0; j < 16; ++j) {
            int k = kc * 32 + frag_k(j, koff);
            float v = (k < D_EMB) ? A[(row0 + m) * (long)D_EMB + k] : 0.0f;
            afr[kc][j] = (bf16_t)v;
        }
    }

    const bf16_t* pb = packedB + (size_t)dir * NT * KC_IN * 512;
    for (int nt = 0; nt < NT; ++nt) {
        const int n = nt * 16 + m;
        const float bias = (n < NG) ? (bih[n] + bhh[n]) : 0.0f;
        v8f acc;
#pragma unroll
        for (int v = 0; v < 8; ++v) acc[v] = bias;

        // Double-buffered B fragments: next load issues before current wmma.
        const bf16_t* pnt = pb + (size_t)nt * KC_IN * 512 + (size_t)lane * 16;
        v16bf bcur = *reinterpret_cast<const v16bf*>(pnt);
#pragma unroll
        for (int kc = 0; kc < KC_IN; ++kc) {
            v16bf bnxt = bcur;
            if (kc + 1 < KC_IN)
                bnxt = *reinterpret_cast<const v16bf*>(pnt + (size_t)(kc + 1) * 512);
            acc = __builtin_amdgcn_wmma_f32_16x16x32_bf16(
                false, afr[kc], false, bcur, (short)0, acc, false, false);
            bcur = bnxt;
        }

#pragma unroll
        for (int v = 0; v < 8; ++v) {
            long row = row0 + v + 8 * half;               // C/D layout: M = v + 8*half
            size_t orow = TRANS ? ((size_t)(row & (W_SENT - 1)) * S_DOC + (size_t)(row >> 7))
                                : (size_t)row;
            size_t off = ((size_t)dir * Mrows + orow) * NPAD + n;
            if (OUT_BF16) ((bf16_t*)outp)[off] = (bf16_t)acc[v];
            else          ((float*)outp)[off]  = acc[v];
        }
    }
}

// ---------------------------------------------------------------------------
// Word-level LSTM recurrence + max-pool. One block = 16 sentences x 1 dir,
// 20 waves, one workgroup per WGP (launch_bounds) for max VGPR budget.
// Each wave owns <=2 N-tiles; its Whh B-fragments (<=10 x 8 VGPRs) stay
// register-resident across all 128 steps. Per k-chunk, ONE A-fragment is
// loaded from LDS and feeds both tiles' accumulators (live set ~104 VGPRs).
// Tile guards are scalar (readfirstlane) so EXEC stays all-ones for WMMA.
// xw is read time-major (contiguous 16x608 bf16 panel per step).
// grid=(64, 2), block=640.
// ---------------------------------------------------------------------------
__global__ void __launch_bounds__(WL_THREADS, 1)
word_lstm_kernel(const bf16_t* __restrict__ xw,     // [2][W][S][NPAD] bf16
                 const bf16_t* __restrict__ pWhh,   // [2][NT][KC_H][512]
                 float* __restrict__ sent_emb) {    // [S][300]
    __shared__ float gbuf[16 * NPAD];                       // 38,912 B
    __shared__ float cbuf[16 * 152];                        //  9,728 B
    __shared__ float mbuf[16 * 152];                        //  9,728 B
    __shared__ __align__(32) bf16_t hfrag[KC_H * 32 * 16];  //  5,120 B

    const int tid  = threadIdx.x;
    const int lane = tid & 31;
    const int wid  = __builtin_amdgcn_readfirstlane(tid >> 5);   // scalar wave id
    const int dir  = blockIdx.y;
    const int sbase = blockIdx.x * 16;
    const int nl = lane & 15;
    const int half = lane >> 4;
    const int nt0 = wid;                 // first tile (always < NT since NT=38 > 19)
    const int nt1 = wid + WL_WAVES;      // second tile, valid iff < NT
    const bool has1 = (nt1 < NT);        // scalar condition

    for (int i = tid; i < KC_H * 32 * 16; i += WL_THREADS) hfrag[i] = (bf16_t)0.0f;
    for (int i = tid; i < 16 * 152; i += WL_THREADS) { cbuf[i] = 0.0f; mbuf[i] = -FLT_MAX; }

    // Preload this wave's Whh B-fragments into registers (persistent).
    const bf16_t* pb = pWhh + (size_t)dir * NT * KC_H * 512;
    v16bf breg0[KC_H], breg1[KC_H];
#pragma unroll
    for (int kc = 0; kc < KC_H; ++kc) {
        breg0[kc] = *reinterpret_cast<const v16bf*>(
            pb + ((size_t)(nt0 * KC_H + kc)) * 512 + (size_t)lane * 16);
        breg1[kc] = has1 ? *reinterpret_cast<const v16bf*>(
            pb + ((size_t)(nt1 * KC_H + kc)) * 512 + (size_t)lane * 16) : breg0[kc];
    }
    __syncthreads();

    const size_t xwdir = (size_t)dir * W_SENT * S_DOC * NPAD;

    for (int s = 0; s < W_SENT; ++s) {
        const int t = dir ? (W_SENT - 1 - s) : s;
        const size_t xwt = xwdir + (size_t)t * S_DOC * NPAD;

        if (has1) {
            // Dual accumulators share each A-fragment as it streams from LDS.
            v8f acc0, acc1;
#pragma unroll
            for (int v = 0; v < 8; ++v) {
                size_t r = xwt + (size_t)(sbase + v + 8 * half) * NPAD;
                acc0[v] = (float)xw[r + nt0 * 16 + nl];
                acc1[v] = (float)xw[r + nt1 * 16 + nl];
            }
#pragma unroll
            for (int kc = 0; kc < KC_H; ++kc) {
                v16bf a = *reinterpret_cast<const v16bf*>(&hfrag[(kc * 32 + lane) * 16]);
                acc0 = __builtin_amdgcn_wmma_f32_16x16x32_bf16(
                    false, a, false, breg0[kc], (short)0, acc0, false, false);
                acc1 = __builtin_amdgcn_wmma_f32_16x16x32_bf16(
                    false, a, false, breg1[kc], (short)0, acc1, false, false);
            }
#pragma unroll
            for (int v = 0; v < 8; ++v) {
                gbuf[(v + 8 * half) * NPAD + nt0 * 16 + nl] = acc0[v];
                gbuf[(v + 8 * half) * NPAD + nt1 * 16 + nl] = acc1[v];
            }
        } else {
            v8f acc0;
#pragma unroll
            for (int v = 0; v < 8; ++v)
                acc0[v] = (float)xw[xwt + (size_t)(sbase + v + 8 * half) * NPAD + nt0 * 16 + nl];
#pragma unroll
            for (int kc = 0; kc < KC_H; ++kc) {
                v16bf a = *reinterpret_cast<const v16bf*>(&hfrag[(kc * 32 + lane) * 16]);
                acc0 = __builtin_amdgcn_wmma_f32_16x16x32_bf16(
                    false, a, false, breg0[kc], (short)0, acc0, false, false);
            }
#pragma unroll
            for (int v = 0; v < 8; ++v)
                gbuf[(v + 8 * half) * NPAD + nt0 * 16 + nl] = acc0[v];
        }
        __syncthreads();

        // Gate math: i,f,g,o blocks of H; write h back in A-fragment order.
        for (int e = tid; e < 16 * 152; e += WL_THREADS) {
            int m = e / 152, h = e % 152;
            if (h < H_HID) {
                float gi = gbuf[m * NPAD + h];
                float gf = gbuf[m * NPAD + H_HID + h];
                float gg = gbuf[m * NPAD + 2 * H_HID + h];
                float go = gbuf[m * NPAD + 3 * H_HID + h];
                float cn = sigf(gf) * cbuf[e] + sigf(gi) * tanhf(gg);
                float hn = sigf(go) * tanhf(cn);
                cbuf[e] = cn;
                mbuf[e] = fmaxf(mbuf[e], hn);
                int kc = h >> 5, kl = h & 31;
                int hb = ((kl & 15) >= 8) ? 1 : 0;
                int j  = (kl & 7) | ((kl >> 4) << 3);
                int ln = (hb << 4) | m;
                hfrag[((kc * 32 + ln) << 4) | j] = (bf16_t)hn;
            }
        }
        __syncthreads();
    }

    for (int e = tid; e < 16 * 152; e += WL_THREADS) {
        int m = e / 152, h = e % 152;
        if (h < H_HID)
            sent_emb[(size_t)(sbase + m) * D_EMB + dir * H_HID + h] = mbuf[e];
    }
}

// ---------------------------------------------------------------------------
// Sentence-level recurrence: M=1, strictly sequential -> VALU. One block/dir.
// ---------------------------------------------------------------------------
__global__ void sent_lstm_kernel(const float* __restrict__ xw2,     // [2][S][NPAD]
                                 const float* __restrict__ Whh_f,
                                 const float* __restrict__ Whh_b,
                                 float* __restrict__ sent_out) {    // [S][300]
    __shared__ float hbuf[152];
    __shared__ float cb[152];
    __shared__ float gb[NG];
    const int tid = threadIdx.x;
    const int dir = blockIdx.x;
    const float* Whh = dir ? Whh_b : Whh_f;
    for (int i = tid; i < 152; i += blockDim.x) { hbuf[i] = 0.0f; cb[i] = 0.0f; }
    __syncthreads();
    for (int st = 0; st < S_DOC; ++st) {
        int t = dir ? (S_DOC - 1 - st) : st;
        for (int n = tid; n < NG; n += blockDim.x) {
            float a = xw2[((size_t)dir * S_DOC + t) * NPAD + n];
            const float* wr = Whh + (size_t)n * H_HID;
            for (int k = 0; k < H_HID; ++k) a += wr[k] * hbuf[k];
            gb[n] = a;
        }
        __syncthreads();
        for (int h = tid; h < H_HID; h += blockDim.x) {
            float cn = sigf(gb[H_HID + h]) * cb[h] + sigf(gb[h]) * tanhf(gb[2 * H_HID + h]);
            float hn = sigf(gb[3 * H_HID + h]) * tanhf(cn);
            cb[h] = cn; hbuf[h] = hn;
            sent_out[(size_t)t * D_EMB + dir * H_HID + h] = hn;
        }
        __syncthreads();
    }
}

// ---------------------------------------------------------------------------
// Head: logits, log-softmax, argmax, NLL, cosine-to-prototype, loss sum.
// One wave per sentence, shfl_xor butterfly reductions. grid=128, block=256.
// ---------------------------------------------------------------------------
__global__ void head_kernel(const float* __restrict__ sent_out,
                            const float* __restrict__ Wout, const float* __restrict__ bout,
                            const float* __restrict__ last_rep,
                            const int* __restrict__ gold,
                            float* __restrict__ out) {
    const int lane = threadIdx.x & 31, wid = threadIdx.x >> 5;
    const int s = blockIdx.x * 8 + wid;
    const float* x = sent_out + (size_t)s * D_EMB;

    float l[C_LBL]; float nx = 0.0f;
#pragma unroll
    for (int c = 0; c < C_LBL; ++c) l[c] = 0.0f;
    for (int j = lane; j < D_EMB; j += 32) {
        float xv = x[j];
        nx += xv * xv;
#pragma unroll
        for (int c = 0; c < C_LBL; ++c) l[c] += xv * Wout[c * D_EMB + j];
    }
#pragma unroll
    for (int c = 0; c < C_LBL; ++c)
        for (int off = 16; off > 0; off >>= 1) l[c] += __shfl_xor(l[c], off);
    for (int off = 16; off > 0; off >>= 1) nx += __shfl_xor(nx, off);
#pragma unroll
    for (int c = 0; c < C_LBL; ++c) l[c] += bout[c];

    int best = 0;
#pragma unroll
    for (int c = 1; c < C_LBL; ++c) if (l[c] > l[best]) best = c;   // first-max
    float mx = l[best];
    float se = 0.0f;
#pragma unroll
    for (int c = 0; c < C_LBL; ++c) se += __expf(l[c] - mx);
    const int g = gold[s];
    float lg = l[0];
#pragma unroll
    for (int c = 1; c < C_LBL; ++c) lg = (c == g) ? l[c] : lg;
    float nll = -(lg - mx - __logf(se));

    const float* p = last_rep + (size_t)best * D_EMB;
    float dp = 0.0f, nb = 0.0f;
    for (int j = lane; j < D_EMB; j += 32) {
        float xv = x[j], pv = p[j];
        dp += xv * pv; nb += pv * pv;
    }
    for (int off = 16; off > 0; off >>= 1) {
        dp += __shfl_xor(dp, off); nb += __shfl_xor(nb, off);
    }
    float sim = dp / (fmaxf(sqrtf(nx), 1e-8f) * fmaxf(sqrtf(nb), 1e-8f));
    float contrib = nll + ((best != g) ? sim : 0.0f);
    if (lane == 0) {
        out[s] = (float)best;
        atomicAdd(&out[S_DOC], contrib);
    }
}

__global__ void zero_loss_kernel(float* out) { out[S_DOC] = 0.0f; }

// ---------------------------------------------------------------------------
extern "C" void kernel_launch(void* const* d_in, const int* in_sizes, int n_in,
                              void* d_out, int out_size, void* d_ws, size_t ws_size,
                              hipStream_t stream) {
    (void)in_sizes; (void)n_in; (void)out_size; (void)ws_size;

    const float* sentences = (const float*)d_in[0];
    const int*   gold      = (const int*)d_in[1];
    const float* Wih_f1 = (const float*)d_in[2],  *Whh_f1 = (const float*)d_in[3];
    const float* bih_f1 = (const float*)d_in[4],  *bhh_f1 = (const float*)d_in[5];
    const float* Wih_b1 = (const float*)d_in[6],  *Whh_b1 = (const float*)d_in[7];
    const float* bih_b1 = (const float*)d_in[8],  *bhh_b1 = (const float*)d_in[9];
    const float* Wih_f2 = (const float*)d_in[10], *Whh_f2 = (const float*)d_in[11];
    const float* bih_f2 = (const float*)d_in[12], *bhh_f2 = (const float*)d_in[13];
    const float* Wih_b2 = (const float*)d_in[14], *Whh_b2 = (const float*)d_in[15];
    const float* bih_b2 = (const float*)d_in[16], *bhh_b2 = (const float*)d_in[17];
    const float* W_out  = (const float*)d_in[18], *b_out  = (const float*)d_in[19];
    const float* last_rep = (const float*)d_in[20];
    float* out = (float*)d_out;

    // Workspace carve-up (all offsets multiples of 256 B).
    char* w = (char*)d_ws;
    bf16_t* XW1 = (bf16_t*)w;   w += (size_t)2 * S_DOC * W_SENT * NPAD * sizeof(bf16_t);
    bf16_t* pWih1 = (bf16_t*)w; w += (size_t)2 * NT * KC_IN * 512 * sizeof(bf16_t);
    bf16_t* pWhh1 = (bf16_t*)w; w += (size_t)2 * NT * KC_H  * 512 * sizeof(bf16_t);
    bf16_t* pWih2 = (bf16_t*)w; w += (size_t)2 * NT * KC_IN * 512 * sizeof(bf16_t);
    float* sent_emb = (float*)w; w += (size_t)S_DOC * D_EMB * sizeof(float);
    float* XW2 = (float*)w;      w += (size_t)2 * S_DOC * NPAD * sizeof(float);
    float* sent_out = (float*)w; w += (size_t)S_DOC * D_EMB * sizeof(float);

    zero_loss_kernel<<<1, 1, 0, stream>>>(out);

    // Pack weights into WMMA-B fragment order (bf16).
    pack_b_kernel<<<dim3(NT, KC_IN), 32, 0, stream>>>(Wih_f1, D_EMB, KC_IN, pWih1);
    pack_b_kernel<<<dim3(NT, KC_IN), 32, 0, stream>>>(Wih_b1, D_EMB, KC_IN, pWih1 + (size_t)NT * KC_IN * 512);
    pack_b_kernel<<<dim3(NT, KC_H),  32, 0, stream>>>(Whh_f1, H_HID, KC_H,  pWhh1);
    pack_b_kernel<<<dim3(NT, KC_H),  32, 0, stream>>>(Whh_b1, H_HID, KC_H,  pWhh1 + (size_t)NT * KC_H * 512);
    pack_b_kernel<<<dim3(NT, KC_IN), 32, 0, stream>>>(Wih_f2, D_EMB, KC_IN, pWih2);
    pack_b_kernel<<<dim3(NT, KC_IN), 32, 0, stream>>>(Wih_b2, D_EMB, KC_IN, pWih2 + (size_t)NT * KC_IN * 512);

    // Word-level input projection, stored time-major bf16: [dir][t][s][608].
    inproj_gemm_kernel<true, true><<<dim3(1024, 1, 2), 256, 0, stream>>>(
        sentences, (long)(S_DOC * W_SENT / 16), (long)(S_DOC * W_SENT),
        pWih1, bih_f1, bhh_f1, bih_b1, bhh_b1, (void*)XW1);

    // Word-level recurrence + max-pool -> sentence embeddings.
    word_lstm_kernel<<<dim3(S_DOC / 16, 2), WL_THREADS, 0, stream>>>(XW1, pWhh1, sent_emb);

    // Sentence-level input projection: [1024 x 300] @ [300 x 600] + bias, f32.
    inproj_gemm_kernel<false, false><<<dim3(8, 1, 2), 256, 0, stream>>>(
        sent_emb, (long)(S_DOC / 16), (long)S_DOC,
        pWih2, bih_f2, bhh_f2, bih_b2, bhh_b2, (void*)XW2);

    // Sentence-level recurrence (sequential, M=1 -> VALU).
    sent_lstm_kernel<<<2, 256, 0, stream>>>(XW2, Whh_f2, Whh_b2, sent_out);

    // Head: logits, argmax, NLL + cosine penalty, loss sum.
    head_kernel<<<S_DOC / 8, 256, 0, stream>>>(sent_out, W_out, b_out, last_rep, gold, out);
}
